// PrototypeLearner_26079041422071
// MI455X (gfx1250) — matrix operand order repaired and checked by reading, not compiled, not run on hardware
//
#include <hip/hip_runtime.h>
#include <hip/hip_bf16.h>
#include <stdint.h>

#define BS   32
#define HW   1024
#define HID  256
#define TOK  768
#define NTOK 2048

typedef __bf16 bf16;
typedef __attribute__((ext_vector_type(16))) __bf16 v16bf;
typedef __attribute__((ext_vector_type(8)))  float  v8f;

static __device__ __forceinline__ bf16 f2bf(float f) {
  union { float f; uint32_t u; } v; v.f = f;
  uint32_t r = (v.u + 0x7FFFu + ((v.u >> 16) & 1u)) >> 16;   // RNE
  uint16_t s = (uint16_t)r;
  bf16 out;
  __builtin_memcpy(&out, &s, 2);
  return out;
}

union FragBF { v16bf v; uint4 u[2]; };

// A fragment (16x32 bf16): lane holds row m = lane&15; hi = lane>>4 selects
// k halves {0..7,16..23} (hi=0) or {8..15,24..31} (hi=1). `row` -> 32 contiguous bf16.
static __device__ __forceinline__ v16bf frag_a_row(const bf16* row, int hi) {
  FragBF f;
  const uint8_t* p = (const uint8_t*)row + (hi ? 16 : 0);
  f.u[0] = *(const uint4*)(p);        // k 0..7   or 8..15
  f.u[1] = *(const uint4*)(p + 32);   // k 16..23 or 24..31
  return f.v;
}

// B fragment (32x16 bf16): lane holds col n = lane&15; hi selects k {0..7,16..23} / {8..15,24..31}.
// `col` -> 32 contiguous bf16 (k-major storage of one column).
static __device__ __forceinline__ v16bf frag_b_col(const bf16* col, int hi) {
  FragBF f;
  const uint8_t* p = (const uint8_t*)col + (hi ? 16 : 0);
  f.u[0] = *(const uint4*)(p);
  f.u[1] = *(const uint4*)(p + 32);
  return f.v;
}

#define WMMA_BF16(A, B, C) \
  __builtin_amdgcn_wmma_f32_16x16x32_bf16(false, (A), false, (B), (short)0, (C), false, false)

// ---------------- K0: codebook norms + bf16 copy ----------------
__global__ void k0_prep(const float* __restrict__ cb, float* __restrict__ cbn,
                        bf16* __restrict__ cbh) {
  int lane = threadIdx.x & 31, wv = threadIdx.x >> 5;
  int j = blockIdx.x * 8 + wv;
  const float* row = cb + (size_t)j * TOK;
  bf16* orow = cbh + (size_t)j * TOK;
  float s = 0.f;
  for (int t = lane; t < TOK; t += 32) { float c = row[t]; s += c * c; orow[t] = f2bf(c); }
  for (int off = 16; off; off >>= 1) s += __shfl_xor(s, off);
  if (lane == 0) cbn[j] = s;
}

// ---------------- K1: P[j,t] = sum_k cb[j,k]*w_pos[t,k] + b_pos[t] ----------------
__global__ void k1_pos(const float* __restrict__ cb, const float* __restrict__ w_pos,
                       const float* __restrict__ b_pos, float* __restrict__ P) {
  __shared__ __align__(16) bf16 As[64][40];
  __shared__ __align__(16) bf16 Bs[64][40];
  int tid = threadIdx.x, lane = tid & 31, wv = tid >> 5;
  int n = lane & 15, hi = lane >> 4;
  int wm = wv & 1, wn = wv >> 1;
  int jbase = blockIdx.x * 64, tbase = blockIdx.y * 64;
  v8f acc0 = {}, acc1 = {};
  for (int k0 = 0; k0 < TOK; k0 += 32) {
    {
      int m = tid >> 2, kc = (tid & 3) * 8;
      const float* sa = cb    + (size_t)(jbase + m) * TOK + k0 + kc;
      const float* sb = w_pos + (size_t)(tbase + m) * TOK + k0 + kc;
#pragma unroll
      for (int e = 0; e < 8; ++e) { As[m][kc + e] = f2bf(sa[e]); Bs[m][kc + e] = f2bf(sb[e]); }
    }
    __syncthreads();
    v16bf fb  = frag_b_col(&Bs[wn * 16 + n][0], hi);
    v16bf fa0 = frag_a_row(&As[wm * 32 + n][0], hi);
    v16bf fa1 = frag_a_row(&As[wm * 32 + 16 + n][0], hi);
    acc0 = WMMA_BF16(fa0, fb, acc0);
    acc1 = WMMA_BF16(fa1, fb, acc1);
    __syncthreads();
  }
  int t = tbase + wn * 16 + n;
  float bias = b_pos[t];
#pragma unroll
  for (int v = 0; v < 8; ++v) {
    int j0 = jbase + wm * 32 + v + hi * 8;
    P[(size_t)j0 * TOK + t]        = acc0[v] + bias;
    P[(size_t)(j0 + 16) * TOK + t] = acc1[v] + bias;
  }
}

// ---------------- K2: xq[b,o,p] = sum_c w_head[o,c]*src[p,b,c] ----------------
__global__ void k2_head(const float* __restrict__ src, const float* __restrict__ w_head,
                        float* __restrict__ xq) {
  __shared__ __align__(16) bf16 As[64][40];
  __shared__ __align__(16) bf16 Bs[64][40];
  int tid = threadIdx.x, lane = tid & 31, wv = tid >> 5;
  int n = lane & 15, hi = lane >> 4;
  int wm = wv & 1, wn = wv >> 1;
  int obase = blockIdx.x * 64, pbase = blockIdx.y * 64, b = blockIdx.z;
  v8f acc0 = {}, acc1 = {};
  for (int k0 = 0; k0 < HID; k0 += 32) {
    {
      int m = tid >> 2, kc = (tid & 3) * 8;
      const float* sa = w_head + (size_t)(obase + m) * HID + k0 + kc;
      const float* sb = src + ((size_t)(pbase + m) * BS + b) * HID + k0 + kc;
#pragma unroll
      for (int e = 0; e < 8; ++e) { As[m][kc + e] = f2bf(sa[e]); Bs[m][kc + e] = f2bf(sb[e]); }
    }
    __syncthreads();
    v16bf fb  = frag_b_col(&Bs[wn * 16 + n][0], hi);
    v16bf fa0 = frag_a_row(&As[wm * 32 + n][0], hi);
    v16bf fa1 = frag_a_row(&As[wm * 32 + 16 + n][0], hi);
    acc0 = WMMA_BF16(fa0, fb, acc0);
    acc1 = WMMA_BF16(fa1, fb, acc1);
    __syncthreads();
  }
  int p = pbase + wn * 16 + n;
  float* out = xq + (size_t)b * TOK * HW;
#pragma unroll
  for (int v = 0; v < 8; ++v) {
    int o0 = obase + wm * 32 + v + hi * 8;
    out[(size_t)o0 * HW + p]        = acc0[v];
    out[(size_t)(o0 + 16) * HW + p] = acc1[v];
  }
}

// ---------------- K3: VQ argmin_j ( ||cb_j||^2 - 2 flat_r . cb_j ) ----------------
__global__ void k3_vq(const float* __restrict__ xq, const bf16* __restrict__ cbh,
                      const float* __restrict__ cbn, int* __restrict__ idx_out) {
  __shared__ __align__(16) bf16 Als[32][776];   // 32 flat rows x 768 (padded)
  __shared__ float redv[8][32];
  __shared__ int   redj[8][32];
  int tid = threadIdx.x, lane = tid & 31, wv = tid >> 5;
  int n = lane & 15, hi = lane >> 4;
  int rbase = blockIdx.x * 32;
  // stage 32 contiguous flat rows (raw view of xq memory), f32 -> bf16
  for (int i = tid; i < 32 * 96; i += 256) {
    int m = i / 96, kc = (i % 96) * 8;
    const float* s = xq + (size_t)(rbase + m) * TOK + kc;
#pragma unroll
    for (int e = 0; e < 8; ++e) Als[m][kc + e] = f2bf(s[e]);
  }
  __syncthreads();
  float bestv[16];
  int   bestj[16];
#pragma unroll
  for (int q = 0; q < 16; ++q) { bestv[q] = 3.4e38f; bestj[q] = 0; }
  for (int jt = 0; jt < 16; ++jt) {          // wave covers j in [wv*256, wv*256+256)
    int j0 = wv * 256 + jt * 16;
    v8f acc0 = {}, acc1 = {};
    const bf16* bcol = cbh + (size_t)(j0 + n) * TOK;
    for (int k0 = 0; k0 < TOK; k0 += 32) {
      v16bf fb  = frag_b_col(bcol + k0, hi);
      v16bf fa0 = frag_a_row(&Als[n][k0], hi);
      v16bf fa1 = frag_a_row(&Als[16 + n][k0], hi);
      acc0 = WMMA_BF16(fa0, fb, acc0);
      acc1 = WMMA_BF16(fa1, fb, acc1);
    }
    int j = j0 + n;
    float cj = cbn[j];
#pragma unroll
    for (int v = 0; v < 8; ++v) {
      float d0 = cj - 2.0f * acc0[v];
      float d1 = cj - 2.0f * acc1[v];
      if (d0 < bestv[v])     { bestv[v] = d0;     bestj[v] = j; }
      if (d1 < bestv[8 + v]) { bestv[8 + v] = d1; bestj[8 + v] = j; }
    }
  }
  // reduce across the 16 lanes of each half (same row set per half)
#pragma unroll
  for (int q = 0; q < 16; ++q) {
    float bv = bestv[q]; int bj = bestj[q];
    for (int off = 1; off < 16; off <<= 1) {
      float ov = __shfl_xor(bv, off);
      int   oj = __shfl_xor(bj, off);
      if (ov < bv || (ov == bv && oj < bj)) { bv = ov; bj = oj; }
    }
    bestv[q] = bv; bestj[q] = bj;
  }
  if (n == 0) {
#pragma unroll
    for (int q = 0; q < 16; ++q) {
      int mt = q >> 3, v = q & 7;
      int row = mt * 16 + hi * 8 + v;        // C layout: M = v + 8*hi
      redv[wv][row] = bestv[q];
      redj[wv][row] = bestj[q];
    }
  }
  __syncthreads();
  if (tid < 32) {
    float bv = redv[0][tid]; int bj = redj[0][tid];
#pragma unroll
    for (int w = 1; w < 8; ++w) {
      float ov = redv[w][tid]; int oj = redj[w][tid];
      if (ov < bv || (ov == bv && oj < bj)) { bv = ov; bj = oj; }
    }
    idx_out[rbase + tid] = bj;
  }
}

// ---------------- K4: gate softmax (2 logits per position) ----------------
__global__ void k4_gate(const float* __restrict__ xq, const float* __restrict__ P,
                        const float* __restrict__ w_gate, const int* __restrict__ idxs,
                        float* __restrict__ s0, float* __restrict__ s1) {
  int lane = threadIdx.x & 31, wv = threadIdx.x >> 5;
  int r = blockIdx.x * 8 + wv;
  int b = r >> 10, p = r & (HW - 1);
  int j = idxs[r];
  const float* e = P + (size_t)j * TOK;
  const float* x = xq + (size_t)b * TOK * HW + p;
  float g0 = 0.f, g1 = 0.f;
  for (int t = lane; t < TOK; t += 32) {
    float ev = e[t];
    float xv = x[(size_t)t * HW];
    g0 += w_gate[t] * ev           + w_gate[TOK + t] * xv;
    g1 += w_gate[2 * TOK + t] * ev + w_gate[3 * TOK + t] * xv;
  }
  for (int off = 16; off; off >>= 1) { g0 += __shfl_xor(g0, off); g1 += __shfl_xor(g1, off); }
  if (lane == 0) {
    float m = fmaxf(g0, g1);
    float e0 = __expf(g0 - m), e1 = __expf(g1 - m);
    float inv = 1.0f / (e0 + e1);
    s0[r] = e0 * inv;
    s1[r] = e1 * inv;
  }
}

// ---------------- K5: out[p,b,o] = sum_t w_tail[o,t] * fused[b,t,p] ----------------
__global__ void k5_tail(const float* __restrict__ xq, const float* __restrict__ P,
                        const float* __restrict__ w_tail, const int* __restrict__ idxs,
                        const float* __restrict__ s0, const float* __restrict__ s1,
                        float* __restrict__ out) {
  __shared__ __align__(16) bf16 As[256][40];
  __shared__ __align__(16) bf16 Bs[32][40];
  int tid = threadIdx.x, lane = tid & 31, wv = tid >> 5;
  int n = lane & 15, hi = lane >> 4;
  int wn = wv & 1, wm = wv >> 1;          // 4(M) x 2(N) wave grid
  int pbase = blockIdx.x * 32, b = blockIdx.y;
  const float* xqb = xq + (size_t)b * TOK * HW;
  v8f zero = {};
  v8f acc[4];
#pragma unroll
  for (int mt = 0; mt < 4; ++mt) acc[mt] = zero;
  // this thread's fused-staging coordinates (fixed across k)
  int pl = tid >> 3, kc = (tid & 7) * 4;
  int pp = pbase + pl;
  int rr = b * HW + pp;
  int jj = idxs[rr];
  float a0 = s0[rr], a1 = s1[rr];
  const float* erow = P + (size_t)jj * TOK;
  for (int k0 = 0; k0 < TOK; k0 += 32) {
    for (int i = tid; i < 1024; i += 256) {          // stage w_tail 256x32
      int m = i >> 2, ac = (i & 3) * 8;
      const float* s = w_tail + (size_t)m * TOK + k0 + ac;
#pragma unroll
      for (int e = 0; e < 8; ++e) As[m][ac + e] = f2bf(s[e]);
    }
#pragma unroll
    for (int q = 0; q < 4; ++q) {                    // stage fused 32pos x 32k
      int t = k0 + kc + q;
      float ev = erow[t];
      float xv = xqb[(size_t)t * HW + pp];
      Bs[pl][kc + q] = f2bf(ev * a0 + xv * a1);
    }
    __syncthreads();
    v16bf fb = frag_b_col(&Bs[wn * 16 + n][0], hi);
#pragma unroll
    for (int mt = 0; mt < 4; ++mt) {
      v16bf fa = frag_a_row(&As[wm * 64 + mt * 16 + n][0], hi);
      acc[mt] = WMMA_BF16(fa, fb, acc[mt]);
    }
    __syncthreads();
  }
  int p = pbase + wn * 16 + n;
#pragma unroll
  for (int mt = 0; mt < 4; ++mt) {
#pragma unroll
    for (int v = 0; v < 8; ++v) {
      int o = wm * 64 + mt * 16 + v + hi * 8;
      out[((size_t)p * BS + b) * HID + o] = acc[mt][v];
    }
  }
}

extern "C" void kernel_launch(void* const* d_in, const int* in_sizes, int n_in,
                              void* d_out, int out_size, void* d_ws, size_t ws_size,
                              hipStream_t stream) {
  (void)in_sizes; (void)n_in; (void)out_size; (void)ws_size;
  const float* src    = (const float*)d_in[0];
  const float* cb     = (const float*)d_in[1];
  const float* w_head = (const float*)d_in[2];
  const float* w_tail = (const float*)d_in[3];
  const float* w_pos  = (const float*)d_in[4];
  const float* b_pos  = (const float*)d_in[5];
  const float* w_gate = (const float*)d_in[6];

  float* out    = (float*)d_out;
  int* idx_out  = (int*)d_out + (size_t)HW * BS * HID;   // indices (int32) after out tensor

  // workspace layout (floats)
  float* ws  = (float*)d_ws;
  float* xq  = ws;                                   // [BS][TOK][HW]   f32
  float* P   = xq  + (size_t)BS * TOK * HW;          // [NTOK][TOK]     f32
  float* cbn = P   + (size_t)NTOK * TOK;             // [NTOK]          f32
  float* s0  = cbn + NTOK;                           // [BS*HW]         f32
  float* s1  = s0  + (size_t)BS * HW;                // [BS*HW]         f32
  bf16*  cbh = (bf16*)(s1 + (size_t)BS * HW);        // [NTOK][TOK]     bf16 (16B aligned)

  k0_prep<<<NTOK / 8, 256, 0, stream>>>(cb, cbn, cbh);
  k1_pos <<<dim3(NTOK / 64, TOK / 64), 256, 0, stream>>>(cb, w_pos, b_pos, P);
  k2_head<<<dim3(TOK / 64, HW / 64, BS), 256, 0, stream>>>(src, w_head, xq);
  k3_vq  <<<(BS * HW) / 32, 256, 0, stream>>>(xq, cbh, cbn, idx_out);
  k4_gate<<<(BS * HW) / 8, 256, 0, stream>>>(xq, P, w_gate, idx_out, s0, s1);
  k5_tail<<<dim3(HW / 32, BS), 256, 0, stream>>>(xq, P, w_tail, idx_out, s0, s1, out);
}
